// TokDctMultiHeadAttention_5428838662762
// MI455X (gfx1250) — matrix operand order repaired and checked by reading, not compiled, not run on hardware
//
#include <hip/hip_runtime.h>
#include <hip/hip_bf16.h>

// Problem constants (match reference)
#define B_   2
#define S_   2048
#define D_   768
#define H_   8
#define HD_  120
#define DK_  960
#define NR_  32        // padded basis rank: 31 real basis fns + 1 zero row
#define TWO_PI 6.2831853071795864769252867665590

typedef __attribute__((ext_vector_type(16))) __bf16 v16bf;
typedef __attribute__((ext_vector_type(8)))  float  v8f;
typedef __attribute__((ext_vector_type(4)))  float  f32x4;
typedef __attribute__((ext_vector_type(4)))  int    i32x4;

// LDS row stride (floats). 16*LDSS*4 = 132,352 B < 320 KB/WGP.
// 2068 % 8 != 0  -> rows v and v+8 hit different bank groups (phase-1 stores);
// 2068 % 64 = 20 -> adjacent rows shifted 20 banks (phase-2 reads);
// 2068 % 4 == 0  -> every row base stays 16B-aligned for b128 LDS access.
#define LDSS 2068

// ---------------------------------------------------------------------------
// K0: build real rfft-lowpass basis. Phi[r][t] (f32, row-major) and
//     PhiT[t][r] (bf16, for WMMA B-operand). r=0: DC; r=2f-1: cos; r=2f: sin;
//     r=31: zero padding row.
// ---------------------------------------------------------------------------
__global__ void build_phi(float* __restrict__ Phi, __bf16* __restrict__ PhiT) {
  int idx = blockIdx.x * blockDim.x + threadIdx.x;
  if (idx >= NR_ * S_) return;
  int r = idx / S_, t = idx % S_;
  double val;
  if (r == 0)            val = 1.0;
  else if (r == NR_ - 1) val = 0.0;
  else {
    int f = (r + 1) >> 1;
    double ang = (TWO_PI * (double)f * (double)t) / (double)S_;
    val = (r & 1) ? cos(ang) : sin(ang);
  }
  float v = (float)val;
  Phi[r * S_ + t]   = v;
  PhiT[t * NR_ + r] = (__bf16)v;
}

// ---------------------------------------------------------------------------
// K1: C[b][r][c] = sum_s X[b][s][c] * Phi[r][s]
// ---------------------------------------------------------------------------
__global__ void basis_project(const float* __restrict__ X,
                              const float* __restrict__ Phi,
                              float* __restrict__ C) {
  int idx = blockIdx.x * blockDim.x + threadIdx.x;
  if (idx >= B_ * NR_ * D_) return;
  int c = idx % D_;
  int r = (idx / D_) % NR_;
  int b = idx / (D_ * NR_);
  const float* x = X + (size_t)b * S_ * D_ + c;   // stride D_ over s -> coalesced in c
  const float* p = Phi + (size_t)r * S_;
  float acc = 0.f;
  for (int s = 0; s < S_; ++s) acc = fmaf(x[(size_t)s * D_], p[s], acc);
  C[idx] = acc;
}

// ---------------------------------------------------------------------------
// K2: A[b][r][dk] = sum_c C[b][r][c] * W[dk][c]  (+ bias folded into DC term)
// ---------------------------------------------------------------------------
__global__ void weight_project(const float* __restrict__ C,
                               const float* __restrict__ W,
                               const float* __restrict__ bias,
                               float* __restrict__ A) {
  int idx = blockIdx.x * blockDim.x + threadIdx.x;
  if (idx >= B_ * NR_ * DK_) return;
  int dk = idx % DK_;
  int r  = (idx / DK_) % NR_;
  int b  = idx / (DK_ * NR_);
  const float* crow = C + (size_t)(b * NR_ + r) * D_;
  const float* wrow = W + (size_t)dk * D_;
  float acc = (r == 0) ? bias[dk] * (float)S_ : 0.f;
  for (int c = 0; c < D_; ++c) acc = fmaf(crow[c], wrow[c], acc);
  A[idx] = acc;
}

// ---------------------------------------------------------------------------
// K3: M[b][h][r][rp] = (g_r*g_rp/sqrt(HD)) * sum_d Aq[b][r][h*HD+d]*Ak[b][rp][h*HD+d]
// ---------------------------------------------------------------------------
__global__ void make_M(const float* __restrict__ Aq,
                       const float* __restrict__ Ak,
                       float* __restrict__ M) {
  int idx = blockIdx.x * blockDim.x + threadIdx.x;
  if (idx >= B_ * H_ * NR_ * NR_) return;
  int rp = idx % NR_;
  int r  = (idx / NR_) % NR_;
  int h  = (idx / (NR_ * NR_)) % H_;
  int b  = idx / (NR_ * NR_ * H_);
  const float* aq = Aq + (size_t)(b * NR_ + r)  * DK_ + h * HD_;
  const float* ak = Ak + (size_t)(b * NR_ + rp) * DK_ + h * HD_;
  float acc = 0.f;
  for (int d = 0; d < HD_; ++d) acc = fmaf(aq[d], ak[d], acc);
  float gr  = ((r  == 0) ? 1.f : 2.f) / (float)S_;
  float grp = ((rp == 0) ? 1.f : 2.f) / (float)S_;
  M[idx] = acc * gr * grp * rsqrtf((float)HD_);
}

// ---------------------------------------------------------------------------
// K4: U[bh][s][rp] = sum_r Phi[r][s] * M[bh][r][rp]   (bf16, WMMA A-operand)
// ---------------------------------------------------------------------------
__global__ void make_U(const float* __restrict__ Phi,
                       const float* __restrict__ M,
                       __bf16* __restrict__ U) {
  int idx = blockIdx.x * blockDim.x + threadIdx.x;
  if (idx >= B_ * H_ * S_ * NR_) return;
  int rp = idx & (NR_ - 1);
  int s  = (idx >> 5) & (S_ - 1);
  int bh = idx >> 16;                       // 5 bits rp + 11 bits s
  const float* m = M + (size_t)bh * NR_ * NR_ + rp;   // stride NR_ over r
  float acc = 0.f;
#pragma unroll
  for (int r = 0; r < NR_; ++r) acc = fmaf(Phi[r * S_ + s], m[r * NR_], acc);
  U[idx] = (__bf16)acc;
}

// ---------------------------------------------------------------------------
// K5: fused scores + mask + softmax.
//   grid = (S/16 row-blocks, B*H). block = 256 threads = 8 wave32.
//   Phase 1: each wave computes a 16x256 strip of scores = U(16x32) x PhiT(32x2048)
//            with ONE v_wmma_f32_16x16x32_bf16 per 16x16 tile -> LDS.
//            Mask slab is prefetched (global_prefetch) to overlap with WMMA.
//   Phase 2: 16 threads per row, 128-bit vector path: NT mask loads, LDS b128,
//            shfl_xor width-16 reductions, NT b128 output stores.
// ---------------------------------------------------------------------------
__global__ __launch_bounds__(256) void scores_softmax(
    const __bf16* __restrict__ U, const __bf16* __restrict__ PhiT,
    const int* __restrict__ mask, float* __restrict__ out) {
  extern __shared__ float sh[];
  const int bh   = blockIdx.y;
  const int b    = bh / H_;
  const int s0   = blockIdx.x * 16;
  const int tid  = threadIdx.x;
  const int lane = tid & 31;
  const int wave = tid >> 5;

  // Prefetch this block's 128 KB mask slab (contiguous 16 rows) while WMMA runs.
  {
    const char* mblk = (const char*)(mask + ((size_t)b * S_ + s0) * S_);
#pragma unroll
    for (int i = 0; i < 4; ++i)
      __builtin_prefetch(mblk + ((size_t)tid + 256 * i) * 128, 0, 0);
  }

  // ---------- phase 1: WMMA ----------
  {
    // A operand, 16-bit 16x32 layout: lanes 0-15 hold row M=lane, K in
    // {0..7, 16..23}; lanes 16-31 hold row M=lane-16, K in {8..15, 24..31}.
    const int mrow = lane & 15;
    const int kA   = (lane < 16) ? 0 : 8;
    const __bf16* up = U + ((size_t)bh * S_ + (s0 + mrow)) * NR_;
    v16bf a;
#pragma unroll
    for (int i = 0; i < 8; ++i) { a[i] = up[kA + i]; a[8 + i] = up[kA + 16 + i]; }

    // B operand, 32x16: lane n<16 holds column n, K in {0..7, 16..23}; lanes 16-31 hold
    // column n-16, K in {8..15, 24..31}.
    const int kB    = (lane < 16) ? 0 : 8;
    const int rbase = (lane < 16) ? 0 : 8;      // C/D row offset for this half-wave
    const int ncol  = lane & 15;

#pragma unroll 4
    for (int tt = 0; tt < 16; ++tt) {
      const int col = wave * 256 + tt * 16 + ncol;
      const __bf16* pp = PhiT + (size_t)col * NR_;
      v16bf bf;
#pragma unroll
      for (int i = 0; i < 8; ++i) { bf[i] = pp[kB + i]; bf[8 + i] = pp[kB + 16 + i]; }
      v8f acc = {};
      acc = __builtin_amdgcn_wmma_f32_16x16x32_bf16(
          /*neg_a=*/false, a, /*neg_b=*/false, bf,
          /*c_mod=*/(short)0, acc, /*reuse_a=*/false, /*reuse_b=*/false);
      // C/D layout: VGPR v -> (row = v + 8*(lane>=16), col = lane%16)
#pragma unroll
      for (int v = 0; v < 8; ++v)
        sh[(rbase + v) * LDSS + col] = acc[v];
    }
  }
  __syncthreads();

  // ---------- phase 2: mask + softmax (128-bit vector path) ----------
  const int row = tid >> 4;          // 0..15
  const int li  = tid & 15;          // 16 threads per row (one wave = 2 rows)
  const int s   = s0 + row;
  const i32x4* mrow4 = (const i32x4*)(mask + ((size_t)b * S_ + s) * S_);
  f32x4*       orow4 = (f32x4*)(out + ((size_t)bh * S_ + s) * S_);
  f32x4*       srow4 = (f32x4*)(sh + row * LDSS);   // row base is 16B aligned

  float mx = -3.4e38f;
#pragma unroll 4
  for (int j = 0; j < S_ / 64; ++j) {            // 32 iterations of 4 floats
    const int q = li + 16 * j;
    f32x4 v = srow4[q];
    i32x4 m = __builtin_nontemporal_load(mrow4 + q);   // single-use stream
    v.x = (m.x == 0) ? -1e9f : v.x;
    v.y = (m.y == 0) ? -1e9f : v.y;
    v.z = (m.z == 0) ? -1e9f : v.z;
    v.w = (m.w == 0) ? -1e9f : v.w;
    srow4[q] = v;
    mx = fmaxf(mx, fmaxf(fmaxf(v.x, v.y), fmaxf(v.z, v.w)));
  }
#pragma unroll
  for (int off = 8; off; off >>= 1) mx = fmaxf(mx, __shfl_xor(mx, off, 16));

  float sum = 0.f;
#pragma unroll 4
  for (int j = 0; j < S_ / 64; ++j) {
    const int q = li + 16 * j;
    f32x4 v = srow4[q];
    f32x4 e;
    e.x = __expf(v.x - mx);
    e.y = __expf(v.y - mx);
    e.z = __expf(v.z - mx);
    e.w = __expf(v.w - mx);
    srow4[q] = e;
    sum += (e.x + e.y) + (e.z + e.w);
  }
#pragma unroll
  for (int off = 8; off; off >>= 1) sum += __shfl_xor(sum, off, 16);
  const float inv = 1.f / sum;

#pragma unroll 4
  for (int j = 0; j < S_ / 64; ++j) {
    const int q = li + 16 * j;
    f32x4 v = srow4[q];
    v.x *= inv; v.y *= inv; v.z *= inv; v.w *= inv;
    __builtin_nontemporal_store(v, orow4 + q);   // streaming 268 MB: keep out of L2
  }
}

// ---------------------------------------------------------------------------
extern "C" void kernel_launch(void* const* d_in, const int* in_sizes, int n_in,
                              void* d_out, int out_size, void* d_ws, size_t ws_size,
                              hipStream_t stream) {
  const float* query = (const float*)d_in[0];
  const float* key   = (const float*)d_in[1];
  const int*   mask  = (const int*)  d_in[2];
  const float* Wq    = (const float*)d_in[3];
  const float* bq    = (const float*)d_in[4];
  const float* Wk    = (const float*)d_in[5];
  const float* bk    = (const float*)d_in[6];
  float* out = (float*)d_out;

  // workspace carve-up (all offsets 256B-aligned; total ~3.3 MB)
  char* ws = (char*)d_ws;
  float*  Phi  = (float*)(ws + 0);         // 32*2048*4      = 262144
  __bf16* PhiT = (__bf16*)(ws + 262144);   // 2048*32*2      = 131072
  float*  Qc   = (float*)(ws + 393216);    // 2*32*768*4     = 196608
  float*  Kc   = (float*)(ws + 589824);    // 196608
  float*  Aq   = (float*)(ws + 786432);    // 2*32*960*4     = 245760
  float*  Ak   = (float*)(ws + 1032192);   // 245760
  float*  Mm   = (float*)(ws + 1277952);   // 2*8*32*32*4    = 65536
  __bf16* U    = (__bf16*)(ws + 1343488);  // 16*2048*32*2   = 2097152

  build_phi<<<(NR_ * S_ + 255) / 256, 256, 0, stream>>>(Phi, PhiT);

  basis_project<<<(B_ * NR_ * D_ + 255) / 256, 256, 0, stream>>>(query, Phi, Qc);
  basis_project<<<(B_ * NR_ * D_ + 255) / 256, 256, 0, stream>>>(key,   Phi, Kc);

  weight_project<<<(B_ * NR_ * DK_ + 255) / 256, 256, 0, stream>>>(Qc, Wq, bq, Aq);
  weight_project<<<(B_ * NR_ * DK_ + 255) / 256, 256, 0, stream>>>(Kc, Wk, bk, Ak);

  make_M<<<(B_ * H_ * NR_ * NR_ + 255) / 256, 256, 0, stream>>>(Aq, Ak, Mm);
  make_U<<<(B_ * H_ * S_ * NR_ + 255) / 256, 256, 0, stream>>>(Phi, Mm, U);

  dim3 grid(S_ / 16, B_ * H_);
  scores_softmax<<<grid, 256, 16 * LDSS * sizeof(float), stream>>>(U, PhiT, mask, out);
}
